// Transition_Interact_79474074845774
// MI455X (gfx1250) — matrix operand order repaired and checked by reading, not compiled, not run on hardware
//
#include <hip/hip_runtime.h>

// ---------------------------------------------------------------------------
// Problem constants (match reference)
// ---------------------------------------------------------------------------
namespace {
constexpr int Bc = 8, Lc = 512, Uc = 24, Dc = 1024, Hc = 16, Tc = 72, DHc = 64;
constexpr int CSKc = 1024, VOCc = 54945;
}

typedef __attribute__((ext_vector_type(16))) __bf16 v16bf;
typedef __attribute__((ext_vector_type(8)))  float  v8f;

union BFrag { v16bf v; unsigned short s[16]; uint4 q[2]; };
union Acc8  { v8f v; float f[8]; };

// Native bf16 convert (lowers to hardware cvt on gfx1250; RNE).
__device__ __forceinline__ unsigned short f2bf(float f) {
  union { __bf16 h; unsigned short s; } u;
  u.h = (__bf16)f;
  return u.s;
}

// ---------------------------------------------------------------------------
// CDNA5 async global->LDS copy (ASYNCcnt path), feature-detected.
// Builtin signature (from clang diagnostic): (v4i AS1*, v4i AS3*, Imm, Imm).
// ---------------------------------------------------------------------------
#if __has_builtin(__builtin_amdgcn_global_load_async_to_lds_b128)
#define HAVE_ASYNC_LDS 1
typedef int v4i __attribute__((vector_size(16)));
typedef __attribute__((address_space(1))) v4i* as1_v4ip;
typedef __attribute__((address_space(3))) v4i* as3_v4ip;
__device__ __forceinline__ void async_copy16(const float* g, float* lds) {
  __builtin_amdgcn_global_load_async_to_lds_b128(
      (as1_v4ip)(uintptr_t)g,
      (as3_v4ip)(unsigned)(uintptr_t)lds, 0, 0);
}
__device__ __forceinline__ void wait_async0() {
#if __has_builtin(__builtin_amdgcn_s_wait_asynccnt)
  __builtin_amdgcn_s_wait_asynccnt(0);
#else
  asm volatile("s_wait_asynccnt 0" ::: "memory");
#endif
}
#endif

// A fragment: 16x32 bf16 tile, row-major source, compile-time lda.
template<int LDA>
__device__ __forceinline__ v16bf load_a_frag(const unsigned short* __restrict__ A,
                                             int m0, int k0, int lane) {
  int m  = m0 + (lane & 15);
  int kl = k0 + ((lane < 16) ? 0 : 16);
  int kh = k0 + ((lane < 16) ? 8 : 24);
  const unsigned short* p = A + (size_t)m * LDA;
  BFrag fr;
  fr.q[0] = *(const uint4*)(p + kl);   // elements 0..7  -> K = kl..kl+7
  fr.q[1] = *(const uint4*)(p + kh);   // elements 8..15 -> K = kh..kh+7
  return fr.v;
}

// ---------------------------------------------------------------------------
// Unified multi-accumulator GEMM:
//   C(M,N) = A_bf16(M,K) * W_f32(K,N) + bias
// One wave owns a 16-wide N stripe and MT vertically-stacked 16-row tiles,
// so each weight element is read once per 16*MT output rows. N, K, MT are
// compile-time -> all B-element strides fold into global_load immediates.
// NCHECK: ragged N (clamp loads, guard stores). NT: stream weights/output
// with non-temporal hints (one-shot 225 MB bow_W must not evict L2).
// ---------------------------------------------------------------------------
template<int MT, int N, int K, bool NCHECK, bool NT>
__global__ __launch_bounds__(256)
void gemm_mt(const unsigned short* __restrict__ A, const float* __restrict__ W,
             const float* __restrict__ bias, float* __restrict__ C, int M) {
  constexpr int NTILES = (N + 15) / 16;
  int lane = threadIdx.x & 31;
  int wid  = (int)((blockIdx.x * blockDim.x + threadIdx.x) >> 5);
  int mgroups = M / (16 * MT);
  if (wid >= NTILES * mgroups) return;            // wave-uniform
  int tg = wid / NTILES, tn = wid % NTILES;
  int ncol = (tn << 4) + (lane & 15);             // true column (for store)
  int n = ncol;
  if (NCHECK) n = (n < N - 1) ? n : (N - 1);      // clamped column (for loads)
  const int kb_lane = (lane < 16) ? 0 : 16;
  const int m0g = tg * 16 * MT;

  v8f acc[MT];
#pragma unroll
  for (int i = 0; i < MT; ++i) acc[i] = (v8f){0.f,0.f,0.f,0.f,0.f,0.f,0.f,0.f};

  const float* Wcol = W + n;
#pragma unroll 2
  for (int k0 = 0; k0 < K; k0 += 32) {
    const float* p = Wcol + (size_t)(k0 + kb_lane) * N;
    if (NT && k0 + 64 < K)                        // stream-ahead prefetch
      __builtin_prefetch(p + (size_t)64 * N, 0, 0);
    BFrag bf;
#pragma unroll
    for (int e = 0; e < 16; ++e) {
      float wv = NT ? __builtin_nontemporal_load(p + (size_t)e * N)
                    : p[(size_t)e * N];
      bf.s[e] = f2bf(wv);
    }
#pragma unroll
    for (int mt = 0; mt < MT; ++mt) {
      v16bf a = load_a_frag<K>(A, m0g + (mt << 4), k0, lane);
      acc[mt] = __builtin_amdgcn_wmma_f32_16x16x32_bf16(false, a, false, bf.v,
                                                        (short)0, acc[mt], false, false);
    }
  }

  if (NCHECK && ncol >= N) return;                // after all WMMAs
  float bv = bias ? bias[ncol] : 0.f;
  int mbase = (lane < 16) ? 0 : 8;
#pragma unroll
  for (int mt = 0; mt < MT; ++mt) {
    Acc8 a; a.v = acc[mt];
#pragma unroll
    for (int v = 0; v < 8; ++v) {
      float* dst = C + (size_t)(m0g + (mt << 4) + mbase + v) * N + ncol;
      if (NT) __builtin_nontemporal_store(a.f[v] + bv, dst);
      else    *dst = a.f[v] + bv;
    }
  }
}

// ---------------------------------------------------------------------------
// Elementwise / prep kernels
// ---------------------------------------------------------------------------
__global__ void gather_cls(const float* __restrict__ hs, const int* __restrict__ idx,
                           float* __restrict__ cls, float* __restrict__ sem_cls_out) {
  int g = blockIdx.x * blockDim.x + threadIdx.x;
  if (g >= Bc * Uc * Dc) return;
  int d = g & (Dc - 1); int bu = g / Dc; int b = bu / Uc;
  int t = idx[bu];
  float v = hs[((size_t)b * Lc + t) * Dc + d];
  cls[g] = v; sem_cls_out[g] = v;
}

__global__ void bn_csk(const float* __restrict__ csk, const float* __restrict__ mu,
                       const float* __restrict__ var, const float* __restrict__ gam,
                       const float* __restrict__ bet, unsigned short* __restrict__ out) {
  int g = blockIdx.x * blockDim.x + threadIdx.x;
  if (g >= Bc * Uc * CSKc) return;
  int c = g & (CSKc - 1);
  float v = (csk[g] - mu[c]) * rsqrtf(var[c] + 1e-5f) * gam[c] + bet[c];
  out[g] = f2bf(v);
}

__global__ void build_trans_in(const float* __restrict__ cls, const float* __restrict__ cskp,
                               unsigned short* __restrict__ xbf) {
  int g = blockIdx.x * blockDim.x + threadIdx.x;
  if (g >= Bc * Tc * Dc) return;
  int d = g & (Dc - 1); int bt = g / Dc; int b = bt / Tc, t = bt % Tc;
  int u = t / 3, j = t % 3;
  size_t src = ((size_t)b * Uc + u) * Dc + d;
  float v = cls[src];
  if (j == 2 && u < Uc - 1) v += cskp[src];
  xbf[g] = f2bf(v);
}

__global__ void to_bf16(const float* __restrict__ in, unsigned short* __restrict__ out, int n) {
  int g = blockIdx.x * blockDim.x + threadIdx.x;
  if (g < n) out[g] = f2bf(in[g]);
}

__global__ void concat_bf(const float* __restrict__ a, const float* __restrict__ b,
                          unsigned short* __restrict__ cat) {
  int g = blockIdx.x * blockDim.x + threadIdx.x;
  if (g >= Bc * Tc * 2 * Dc) return;
  int c = g & (2 * Dc - 1); int bt = g / (2 * Dc);
  float v = (c < Dc) ? a[(size_t)bt * Dc + c] : b[(size_t)bt * Dc + (c - Dc)];
  cat[g] = f2bf(v);
}

__global__ void gate_mix(const float* __restrict__ gpre, const float* __restrict__ t,
                         const float* __restrict__ it, float* __restrict__ out) {
  int g = blockIdx.x * blockDim.x + threadIdx.x;
  if (g >= Bc * Tc * Dc) return;
  float s = 1.f / (1.f + __expf(-gpre[g]));
  out[g] = (1.f - s) * it[g] + s * t[g];
}

__global__ void slice_outputs(const float* __restrict__ outs, const float* __restrict__ cls,
                              float* __restrict__ emo_out, float* __restrict__ last_stra,
                              float* __restrict__ last_delta, float* __restrict__ stra_buf,
                              unsigned short* __restrict__ delta_bf) {
  int g = blockIdx.x * blockDim.x + threadIdx.x;
  if (g >= Bc * Uc * Dc) return;
  int d = g & (Dc - 1); int bu = g / Dc; int b = bu / Uc, u = bu % Uc;
  size_t base = ((size_t)b * Tc + 3 * u) * Dc + d;
  float sem  = outs[base];
  float stra = outs[base + Dc];
  float emo  = outs[base + 2 * Dc];
  emo_out[g]  = emo;
  stra_buf[g] = stra;
  float delta = sem - cls[g];
  delta_bf[g] = f2bf(delta);
  if (u == Uc - 1) { last_stra[b * Dc + d] = stra; last_delta[b * Dc + d] = delta; }
}

__global__ void head_logits(const float* __restrict__ stra, const float* __restrict__ emo,
                            const float* __restrict__ strW, const float* __restrict__ strB,
                            const float* __restrict__ emoW, const float* __restrict__ emoB,
                            float* __restrict__ str_out, float* __restrict__ emo_out) {
  int g = blockIdx.x * blockDim.x + threadIdx.x;
  int o = g & 15; int bu = g >> 4;
  if (bu >= Bc * Uc) return;
  if (o < 8) {
    float s = strB[o];
    const float* x = stra + (size_t)bu * Dc;
    for (int d = 0; d < Dc; ++d) s += x[d] * strW[d * 8 + o];
    str_out[bu * 8 + o] = s;
  } else if (o < 14) {
    int oo = o - 8;
    float s = emoB[oo];
    const float* x = emo + (size_t)bu * Dc;
    for (int d = 0; d < Dc; ++d) s += x[d] * emoW[d * 6 + oo];
    emo_out[bu * 6 + oo] = s;
  }
}

// cumulative edge-type histogram along i:  cnt[b,i,k,e] = #{i'<=i : et[b,i',k]==e}
__global__ void cum_counts(const int* __restrict__ et, float* __restrict__ cnt) {
  int g = blockIdx.x * blockDim.x + threadIdx.x;
  if (g >= Bc * Tc) return;
  int b = g / Tc, k = g % Tc;
  float c[8] = {0,0,0,0,0,0,0,0};
  for (int i = 0; i < Tc; ++i) {
    int e = et[((size_t)b * Tc + i) * Tc + k] & 7;
#pragma unroll
    for (int j = 0; j < 8; ++j) c[j] += (e == j) ? 1.f : 0.f;
    float* dst = cnt + ((((size_t)b * Tc + i) * Tc + k) << 3);
#pragma unroll
    for (int j = 0; j < 8; ++j) dst[j] = c[j];
  }
}

// ---------------------------------------------------------------------------
// Attention core: one block per (b,h); K/V staged in LDS (padded rows) via
// async global->LDS when available.
// scores[i,k] = (A_i . K_k + sum_e cnt[b,i,k,e] * (A_i . rel_e)) / 8, masked,
// softmax, ctx = probs @ V.  A_i = Q_i + sum_e cnt[b,i,i,e] * rel_e.
// ---------------------------------------------------------------------------
__global__ __launch_bounds__(128)
void attn_core(const float* __restrict__ Qb, const float* __restrict__ Kb,
               const float* __restrict__ Vb, const float* __restrict__ counts,
               const int* __restrict__ graph, const float* __restrict__ rel,
               float* __restrict__ Out) {
  int b = (int)blockIdx.x >> 4;
  int h = (int)blockIdx.x & 15;
  int tid = threadIdx.x, wave = tid >> 5, lane = tid & 31;

  __shared__ float Ks[Tc][DHc + 4];   // pad 4 floats: rows stay 16B-aligned,
  __shared__ float Vs[Tc][DHc + 4];   // banks still skewed (68 % 64 == 4)
  __shared__ float Rs[8][DHc];
  __shared__ float Pb[4][Tc];

#if defined(HAVE_ASYNC_LDS)
  for (int idx = tid; idx < Tc * (DHc / 4); idx += 128) {
    int t = idx / (DHc / 4), c4 = (idx % (DHc / 4)) << 2;
    size_t off = ((size_t)b * Tc + t) * Dc + h * DHc + c4;
    async_copy16(Kb + off, &Ks[t][c4]);
    async_copy16(Vb + off, &Vs[t][c4]);
  }
#else
  for (int idx = tid; idx < Tc * DHc; idx += 128) {
    int t = idx / DHc, d = idx % DHc;
    size_t off = ((size_t)b * Tc + t) * Dc + h * DHc + d;
    Ks[t][d] = Kb[off];
    Vs[t][d] = Vb[off];
  }
#endif
  for (int idx = tid; idx < 8 * DHc; idx += 128)
    Rs[idx / DHc][idx % DHc] = rel[(idx / DHc) * Dc + h * DHc + (idx % DHc)];
#if defined(HAVE_ASYNC_LDS)
  wait_async0();
#endif
  __syncthreads();

  for (int i = wave; i < Tc; i += 4) {
    const float* cII = counts + ((((size_t)b * Tc + i) * Tc + i) << 3);
    size_t qoff = ((size_t)b * Tc + i) * Dc + h * DHc;
    float a0 = Qb[qoff + lane];
    float a1 = Qb[qoff + lane + 32];
#pragma unroll
    for (int e = 0; e < 8; ++e) {
      float c = cII[e];
      a0 += c * Rs[e][lane];
      a1 += c * Rs[e][lane + 32];
    }
    // P[e] = A_i . rel_e  (wave reduction)
    float P[8];
#pragma unroll
    for (int e = 0; e < 8; ++e) {
      float p = a0 * Rs[e][lane] + a1 * Rs[e][lane + 32];
#pragma unroll
      for (int off = 16; off > 0; off >>= 1) p += __shfl_xor(p, off, 32);
      P[e] = p;
    }
    // scores: lane owns k = lane, lane+32, lane+64(<Tc)
    float s[3];
#pragma unroll
    for (int kk = 0; kk < 3; ++kk) {
      int k = lane + kk * 32;
      s[kk] = -3.0e38f;
      if (k < Tc) {
        float acc = 0.f;
        for (int d = 0; d < 32; ++d) {
          float av0 = __shfl(a0, d, 32);
          float av1 = __shfl(a1, d, 32);
          acc += av0 * Ks[k][d] + av1 * Ks[k][d + 32];
        }
        const float* cik = counts + ((((size_t)b * Tc + i) * Tc + k) << 3);
        float t2 = 0.f;
#pragma unroll
        for (int e = 0; e < 8; ++e) t2 += cik[e] * P[e];
        float sv = (acc + t2) * 0.125f;
        int m = graph[((size_t)b * Tc + i) * Tc + k];
        s[kk] = (m == 0) ? -1.0e9f : sv;
      }
    }
    float mx = fmaxf(fmaxf(s[0], s[1]), s[2]);
#pragma unroll
    for (int off = 16; off > 0; off >>= 1) mx = fmaxf(mx, __shfl_xor(mx, off, 32));
    float ex[3]; float sum = 0.f;
#pragma unroll
    for (int kk = 0; kk < 3; ++kk) {
      int k = lane + kk * 32;
      ex[kk] = (k < Tc) ? __expf(s[kk] - mx) : 0.f;
      sum += ex[kk];
    }
#pragma unroll
    for (int off = 16; off > 0; off >>= 1) sum += __shfl_xor(sum, off, 32);
    float inv = 1.f / sum;
#pragma unroll
    for (int kk = 0; kk < 3; ++kk) {
      int k = lane + kk * 32;
      if (k < Tc) Pb[wave][k] = ex[kk] * inv;
    }
    __syncthreads();                     // equal trip count in all 4 waves
    float c0 = 0.f, c1 = 0.f;
    for (int k = 0; k < Tc; ++k) {
      float p = Pb[wave][k];
      c0 += p * Vs[k][lane];
      c1 += p * Vs[k][lane + 32];
    }
    Out[qoff + lane] = c0;
    Out[qoff + lane + 32] = c1;
    __syncthreads();
  }
}

// ---------------------------------------------------------------------------
// Host launcher
// ---------------------------------------------------------------------------
extern "C" void kernel_launch(void* const* d_in, const int* in_sizes, int n_in,
                              void* d_out, int out_size, void* d_ws, size_t ws_size,
                              hipStream_t stream) {
  const float* hs     = (const float*)d_in[0];
  const float* emoCsk = (const float*)d_in[1];
  const float* Wq = (const float*)d_in[2];  const float* bq = (const float*)d_in[3];
  const float* Wk = (const float*)d_in[4];  const float* bk = (const float*)d_in[5];
  const float* Wv = (const float*)d_in[6];  const float* bv = (const float*)d_in[7];
  const float* rel = (const float*)d_in[8];
  const float* fW  = (const float*)d_in[9]; const float* fb = (const float*)d_in[10];
  const float* bnG = (const float*)d_in[11]; const float* bnB = (const float*)d_in[12];
  const float* bnM = (const float*)d_in[13]; const float* bnV = (const float*)d_in[14];
  const float* cskW = (const float*)d_in[15]; const float* cskB = (const float*)d_in[16];
  const float* emoW = (const float*)d_in[17]; const float* emoB = (const float*)d_in[18];
  const float* strW = (const float*)d_in[19]; const float* strB = (const float*)d_in[20];
  const float* bowW = (const float*)d_in[21];
  const int* clsIdx = (const int*)d_in[22];
  const int* tGraph = (const int*)d_in[23];
  const int* iGraph = (const int*)d_in[24];
  const int* tEdge  = (const int*)d_in[25];
  const int* iEdge  = (const int*)d_in[26];

  float* out = (float*)d_out;
  float* o_last_stra  = out;              // 8192
  float* o_emo        = out + 8192;       // 196608
  float* o_last_delta = out + 204800;     // 8192
  float* o_semcls     = out + 212992;     // 196608
  float* o_strlog     = out + 409600;     // 1536
  float* o_emolog     = out + 411136;     // 1152
  float* o_bow        = out + 412288;     // 10549440

  char* w = (char*)d_ws;
  auto take = [&](size_t bytes) -> void* {
    void* p = (void*)w; w += ((bytes + 255) & ~(size_t)255); return p;
  };
  const size_t BUD = (size_t)Bc * Uc * Dc;   // 196608
  const size_t BTD = (size_t)Bc * Tc * Dc;   // 589824

  float*          cls   = (float*)take(4 * BUD);
  unsigned short* cskbf = (unsigned short*)take(2 * (size_t)Bc * Uc * CSKc);
  float*          cskp  = (float*)take(4 * BUD);
  unsigned short* xbf   = (unsigned short*)take(2 * BTD);
  float* q1 = (float*)take(4 * BTD);
  float* k1 = (float*)take(4 * BTD);
  float* v1 = (float*)take(4 * BTD);
  float* cntT = (float*)take(4ull * Bc * Tc * Tc * 8);
  float* cntI = (float*)take(4ull * Bc * Tc * Tc * 8);
  float* att1 = (float*)take(4 * BTD);
  unsigned short* a1bf = (unsigned short*)take(2 * BTD);
  float* att2 = (float*)take(4 * BTD);
  unsigned short* catbf = (unsigned short*)take(2 * BTD * 2);
  float* gatep = (float*)take(4 * BTD);
  float* outs  = (float*)take(4 * BTD);
  float* strab = (float*)take(4 * BUD);
  unsigned short* delbf = (unsigned short*)take(2 * BUD);

  auto cdiv = [](int a, int b) { return (a + b - 1) / b; };
  // waves = ntiles(N) * (M / (16*MT)); blocks of 256 = 8 waves
  auto blocks_for = [&](int M, int N, int MT) {
    int waves = ((N + 15) / 16) * (M / (16 * MT));
    return cdiv(waves * 32, 256);
  };

  // 1) gather CLS + emit sem_cls_embs
  gather_cls<<<cdiv((int)BUD, 256), 256, 0, stream>>>(hs, clsIdx, cls, o_semcls);
  // 2) batchnorm CSK -> bf16
  bn_csk<<<cdiv(Bc * Uc * CSKc, 256), 256, 0, stream>>>(emoCsk, bnM, bnV, bnG, bnB, cskbf);
  // 3) csk_proj = csk_norm @ csk_W + csk_b   (192 x 1024 x 1024)
  gemm_mt<4, Dc, CSKc, false, false>
      <<<blocks_for(192, Dc, 4), 256, 0, stream>>>(cskbf, cskW, cskB, cskp, 192);
  // 4) trans_in assembly -> bf16
  build_trans_in<<<cdiv((int)BTD, 256), 256, 0, stream>>>(cls, cskp, xbf);
  // 5) layer-1 QKV  (576 x 1024 x 1024 each)
  gemm_mt<4, Dc, Dc, false, false>
      <<<blocks_for(Bc * Tc, Dc, 4), 256, 0, stream>>>(xbf, Wq, bq, q1, Bc * Tc);
  gemm_mt<4, Dc, Dc, false, false>
      <<<blocks_for(Bc * Tc, Dc, 4), 256, 0, stream>>>(xbf, Wk, bk, k1, Bc * Tc);
  gemm_mt<4, Dc, Dc, false, false>
      <<<blocks_for(Bc * Tc, Dc, 4), 256, 0, stream>>>(xbf, Wv, bv, v1, Bc * Tc);
  // 6) cumulative edge-type histograms
  cum_counts<<<cdiv(Bc * Tc, 256), 256, 0, stream>>>(tEdge, cntT);
  cum_counts<<<cdiv(Bc * Tc, 256), 256, 0, stream>>>(iEdge, cntI);
  // 7) layer-1 attention
  attn_core<<<Bc * Hc, 128, 0, stream>>>(q1, k1, v1, cntT, tGraph, rel, att1);
  // 8) layer-2 QKV on trans_out
  to_bf16<<<cdiv((int)BTD, 256), 256, 0, stream>>>(att1, a1bf, (int)BTD);
  gemm_mt<4, Dc, Dc, false, false>
      <<<blocks_for(Bc * Tc, Dc, 4), 256, 0, stream>>>(a1bf, Wq, bq, q1, Bc * Tc);
  gemm_mt<4, Dc, Dc, false, false>
      <<<blocks_for(Bc * Tc, Dc, 4), 256, 0, stream>>>(a1bf, Wk, bk, k1, Bc * Tc);
  gemm_mt<4, Dc, Dc, false, false>
      <<<blocks_for(Bc * Tc, Dc, 4), 256, 0, stream>>>(a1bf, Wv, bv, v1, Bc * Tc);
  // 9) layer-2 attention
  attn_core<<<Bc * Hc, 128, 0, stream>>>(q1, k1, v1, cntI, iGraph, rel, att2);
  // 10) fusion gate  (576 x 1024, K=2048)
  concat_bf<<<cdiv((int)(BTD * 2), 256), 256, 0, stream>>>(att1, att2, catbf);
  gemm_mt<4, Dc, 2 * Dc, false, false>
      <<<blocks_for(Bc * Tc, Dc, 4), 256, 0, stream>>>(catbf, fW, fb, gatep, Bc * Tc);
  gate_mix<<<cdiv((int)BTD, 256), 256, 0, stream>>>(gatep, att1, att2, outs);
  // 11) slice outputs, delta, last_* heads
  slice_outputs<<<cdiv((int)BUD, 256), 256, 0, stream>>>(outs, cls, o_emo, o_last_stra,
                                                         o_last_delta, strab, delbf);
  // 12) small logit heads
  head_logits<<<cdiv(Bc * Uc * 16, 256), 256, 0, stream>>>(strab, o_emo, strW, strB,
                                                           emoW, emoB, o_strlog, o_emolog);
  // 13) bow_logits = delta @ bow_W  (dominant: 225 MB weight stream, NT path,
  //     MT=12 -> each weight element read exactly once for all 192 rows)
  gemm_mt<12, VOCc, Dc, true, true>
      <<<blocks_for(192, VOCc, 12), 256, 0, stream>>>(delbf, bowW, nullptr, o_bow, 192);
}
